// MHA_84181359002225
// MI455X (gfx1250) — matrix operand order repaired and checked by reading, not compiled, not run on hardware
//
#include <hip/hip_runtime.h>
#include <hip/hip_bf16.h>

// ---------------------------------------------------------------------------
// MHA forward for MI455X (gfx1250, wave32, WMMA).
// All matmuls run on v_wmma_f32_16x16x32_bf16 (bf16 in, f32 accumulate).
// GEMM staging uses GLOBAL_LOAD_ASYNC_TO_LDS_B128 (ASYNCcnt) when the
// toolchain exposes the builtin, with a sync global->reg->LDS fallback.
// ---------------------------------------------------------------------------

typedef __attribute__((ext_vector_type(16))) __bf16        v16bf;
typedef __attribute__((ext_vector_type(8)))  float         v8f;
typedef __attribute__((ext_vector_type(8)))  unsigned int  v8u;
typedef int v4i __attribute__((vector_size(16)));   // matches builtin param

#define SEQ   2048
#define BATCH 2
#define DMODEL 1024
#define HEADS 16
#define DH    64
#define MROWS (BATCH*SEQ)      // 4096

// ---- CDNA5 async memory->LDS path (guarded; fallback is synchronous) ------
#define ASYNC_OK 0
#if defined(__has_builtin)
#  if __has_builtin(__builtin_amdgcn_global_load_async_to_lds_b128)
#    undef ASYNC_OK
#    define ASYNC_OK 1
#  endif
#endif

__device__ __forceinline__ void cp16_to_lds(unsigned short* l,
                                            const unsigned short* g) {
#if ASYNC_OK
    __builtin_amdgcn_global_load_async_to_lds_b128(
        (__attribute__((address_space(1))) v4i*)g,
        (__attribute__((address_space(3))) v4i*)l, 0, 0);
#else
    *(uint4*)l = *(const uint4*)g;
#endif
}

__device__ __forceinline__ void async_wait0() {
#if ASYNC_OK
#  if __has_builtin(__builtin_amdgcn_s_wait_asynccnt)
    __builtin_amdgcn_s_wait_asynccnt(0);
#  else
    asm volatile("s_wait_asynccnt 0x0" ::: "memory");
#  endif
#endif
}

// ---------------------------------------------------------------------------
__device__ __forceinline__ unsigned short f2bf(float f) {
    unsigned int u = __builtin_bit_cast(unsigned int, f);
    unsigned int r = 0x7FFFu + ((u >> 16) & 1u);   // round-to-nearest-even
    return (unsigned short)((u + r) >> 16);
}

__device__ __forceinline__ v8f zero8() {
    v8f z = {0.f,0.f,0.f,0.f,0.f,0.f,0.f,0.f};
    return z;
}

// Build a 16-element bf16 fragment from two 16-byte chunks (elements 0..7 from
// p0, 8..15 from p1). Works for LDS or global pointers (addrspace inferred).
__device__ __forceinline__ v16bf make_frag(const unsigned short* p0,
                                           const unsigned short* p1) {
    uint4 a = *(const uint4*)p0;
    uint4 b = *(const uint4*)p1;
    v8u t;
    t[0]=a.x; t[1]=a.y; t[2]=a.z; t[3]=a.w;
    t[4]=b.x; t[5]=b.y; t[6]=b.z; t[7]=b.w;
    return __builtin_bit_cast(v16bf, t);
}

// A-operand (16x32, M x K): lane<16 holds row=lane, K {0..7,16..23};
// lane>=16 holds row=lane-16, K {8..15,24..31}. base = row start, elems.
__device__ __forceinline__ v16bf frag_a(const unsigned short* row, int hi) {
    return make_frag(row + hi*8, row + hi*8 + 16);
}
// B-operand (32x16, K x N): lane<16 holds col=lane, K {0..7,16..23};
// lane>=16 holds col=lane-16, K {8..15,24..31}. base = per-column contiguous K run.
__device__ __forceinline__ v16bf frag_b(const unsigned short* col, int hi) {
    return make_frag(col + hi*8, col + hi*8 + 16);
}

__device__ __forceinline__ v8f wmma_bf16(v16bf a, v16bf b, v8f c) {
    return __builtin_amdgcn_wmma_f32_16x16x32_bf16(
        false, a, false, b, (short)0, c, false, false);
}

// ------------------------------- converters --------------------------------
__global__ void k_cvt_bf16(const float* __restrict__ src,
                           unsigned short* __restrict__ dst, int n) {
    int i = (blockIdx.x * blockDim.x + threadIdx.x) * 4;
    if (i < n) {
        float4 f = *(const float4*)(src + i);
        unsigned int lo = (unsigned)f2bf(f.x) | ((unsigned)f2bf(f.y) << 16);
        unsigned int hh = (unsigned)f2bf(f.z) | ((unsigned)f2bf(f.w) << 16);
        *(uint2*)(dst + i) = make_uint2(lo, hh);
    }
}

// W [K=1024][N=1024] fp32 -> WT [N][K] bf16
__global__ void k_transpose_cvt(const float* __restrict__ w,
                                unsigned short* __restrict__ wt) {
    int idx = blockIdx.x * blockDim.x + threadIdx.x;
    int k = idx & (DMODEL - 1);
    int n = idx >> 10;
    wt[(size_t)n * DMODEL + k] = f2bf(w[(size_t)k * DMODEL + n]);
}

// ------------------------------- tiled GEMM --------------------------------
// C[M,N] = A[M,K] @ Bt[N,K]^T + bias ; MODE 0: bf16 row-major,
// 1: bf16 transposed (out[n*M+m]), 2: fp32 row-major * mask[m].
// Double-buffered LDS; tile t+1 staged (async when available) under tile t's
// WMMAs; one barrier per K-step.
#define GM 128
#define GN 128
#define GK 32
#define LSTR 40   // LDS row stride (elements); 80B keeps b128 loads bank-clean

template<int MODE>
__global__ __launch_bounds__(256)
void k_gemm_bf16(const unsigned short* __restrict__ A,
                 const unsigned short* __restrict__ Bt,
                 const float* __restrict__ bias,
                 void* __restrict__ outp,
                 const float* __restrict__ mask,
                 int Mdim, int Kd, int Ndim) {
    __shared__ __align__(16) unsigned short sA[2][GM * LSTR];
    __shared__ __align__(16) unsigned short sB[2][GM * LSTR];

    const int tid  = threadIdx.x;
    const int lane = tid & 31;
    const int wave = tid >> 5;        // 0..7
    const int mw   = wave & 1;        // 2 wave-rows of 64
    const int nw   = wave >> 1;       // 4 wave-cols of 32
    const int m0   = blockIdx.y * GM;
    const int n0   = blockIdx.x * GN;
    const int lh   = lane & 15;
    const int hi   = lane >> 4;

    // per-thread staging coordinates: two b128 chunks per tile per thread
    const int c0   = tid * 2;
    const int row0 = c0 >> 2,        kc0 = (c0 & 3) * 8;
    const int row1 = (c0 + 1) >> 2,  kc1 = ((c0 + 1) & 3) * 8;
    const unsigned short* gA0 = A  + (size_t)(m0 + row0) * Kd + kc0;
    const unsigned short* gA1 = A  + (size_t)(m0 + row1) * Kd + kc1;
    const unsigned short* gB0 = Bt + (size_t)(n0 + row0) * Kd + kc0;
    const unsigned short* gB1 = Bt + (size_t)(n0 + row1) * Kd + kc1;

    auto stage = [&](int buf, int k0) {
        cp16_to_lds(&sA[buf][row0 * LSTR + kc0], gA0 + k0);
        cp16_to_lds(&sA[buf][row1 * LSTR + kc1], gA1 + k0);
        cp16_to_lds(&sB[buf][row0 * LSTR + kc0], gB0 + k0);
        cp16_to_lds(&sB[buf][row1 * LSTR + kc1], gB1 + k0);
    };

    v8f acc[4][2];
    #pragma unroll
    for (int i = 0; i < 4; i++)
        #pragma unroll
        for (int j = 0; j < 2; j++) acc[i][j] = zero8();

    const int nk = Kd / GK;
    stage(0, 0);
    async_wait0();
    __syncthreads();

    for (int kt = 0; kt < nk; kt++) {
        const int buf = kt & 1;
        if (kt + 1 < nk) stage(buf ^ 1, (kt + 1) * GK);

        v16bf af[4], bfr[2];
        #pragma unroll
        for (int mf = 0; mf < 4; mf++)
            af[mf] = frag_a(&sA[buf][(mw * 64 + mf * 16 + lh) * LSTR], hi);
        #pragma unroll
        for (int nf = 0; nf < 2; nf++)
            bfr[nf] = frag_b(&sB[buf][(nw * 32 + nf * 16 + lh) * LSTR], hi);

        #pragma unroll
        for (int mf = 0; mf < 4; mf++)
            #pragma unroll
            for (int nf = 0; nf < 2; nf++)
                acc[mf][nf] = wmma_bf16(af[mf], bfr[nf], acc[mf][nf]);

        async_wait0();
        __syncthreads();
    }

    // epilogue: C frag layout -> VGPR r = row (m + r + hi*8), lane = col n
    const int mbase = m0 + mw * 64;
    const int nbase = n0 + nw * 32;
    #pragma unroll
    for (int nf = 0; nf < 2; nf++) {
        const int n  = nbase + nf * 16 + lh;
        const float bv = bias[n];
        #pragma unroll
        for (int mf = 0; mf < 4; mf++) {
            if constexpr (MODE == 1) {
                // transposed bf16: out[n*Mdim + m]; r=0..7 are 8 consecutive m
                int mrow0 = mbase + mf * 16 + hi * 8;
                unsigned short tmp[8];
                #pragma unroll
                for (int r = 0; r < 8; r++) tmp[r] = f2bf(acc[mf][nf][r] + bv);
                uint4 pk;
                pk.x = (unsigned)tmp[0] | ((unsigned)tmp[1] << 16);
                pk.y = (unsigned)tmp[2] | ((unsigned)tmp[3] << 16);
                pk.z = (unsigned)tmp[4] | ((unsigned)tmp[5] << 16);
                pk.w = (unsigned)tmp[6] | ((unsigned)tmp[7] << 16);
                *(uint4*)((unsigned short*)outp + (size_t)n * Mdim + mrow0) = pk;
            } else if constexpr (MODE == 2) {
                #pragma unroll
                for (int r = 0; r < 8; r++) {
                    int m = mbase + mf * 16 + hi * 8 + r;
                    ((float*)outp)[(size_t)m * Ndim + n] =
                        (acc[mf][nf][r] + bv) * mask[m];
                }
            } else {
                #pragma unroll
                for (int r = 0; r < 8; r++) {
                    int m = mbase + mf * 16 + hi * 8 + r;
                    ((unsigned short*)outp)[(size_t)m * Ndim + n] =
                        f2bf(acc[mf][nf][r] + bv);
                }
            }
        }
    }
}

// ------------------------------- attention ---------------------------------
// grid: (S/64, B*H). 4 waves; each wave owns 16 q-rows, flash-style over 32
// key tiles of 64. Q'/K' are [4096][1024] bf16 row-major; V is stored
// transposed VT[h*64+d][b*S+s] so the PV B-operand reads contiguous key runs.
// Q'/K'/VT (~24 MB bf16) are fully L2-resident (192 MB), so B-fragments load
// straight from global with no LDS staging.
#define PSTR 72   // per-wave P scratch row stride (elements)

__global__ __launch_bounds__(128)
void k_attention(const unsigned short* __restrict__ Qp,
                 const unsigned short* __restrict__ Kp,
                 const unsigned short* __restrict__ VT,
                 const float* __restrict__ mask,   // [B*S]
                 unsigned short* __restrict__ ctx) // [4096][1024]
{
    __shared__ __align__(16) unsigned short sP[4][16 * PSTR];

    const int lane = threadIdx.x & 31;
    const int wave = threadIdx.x >> 5;   // 0..3
    const int lh = lane & 15, hi = lane >> 4;
    const int qt = blockIdx.x;           // q tile (64 rows)
    const int bh = blockIdx.y;
    const int b  = bh >> 4, h = bh & 15;
    const int q0 = qt * 64 + wave * 16;
    const size_t rowQ = (size_t)b * SEQ + q0;
    unsigned short* myP = &sP[wave][0];

    // Q fragments for this wave's 16 rows (held in registers all iterations)
    v16bf aq[2];
    {
        const unsigned short* qrow = Qp + (rowQ + lh) * DMODEL + h * DH;
        aq[0] = frag_a(qrow, hi);
        aq[1] = frag_a(qrow + 32, hi);
    }

    v8f o[4];
    #pragma unroll
    for (int nf = 0; nf < 4; nf++) o[nf] = zero8();
    float rmax[8], rsum[8];
    #pragma unroll
    for (int r = 0; r < 8; r++) { rmax[r] = -1e30f; rsum[r] = 0.f; }

    const float scale = 0.03125f;  // 1/sqrt(1024)

    for (int kt = 0; kt < SEQ / 64; kt++) {
        const int kbase = kt * 64;

        // ---- scores S = Q K^T : B-operand straight from K' rows ----
        v8f s[4];
        #pragma unroll
        for (int nf = 0; nf < 4; nf++) s[nf] = zero8();
        #pragma unroll
        for (int kf = 0; kf < 2; kf++) {
            #pragma unroll
            for (int nf = 0; nf < 4; nf++) {
                const unsigned short* krow =
                    Kp + ((size_t)b * SEQ + kbase + nf * 16 + lh) * DMODEL
                       + h * DH + kf * 32;
                s[nf] = wmma_bf16(aq[kf], frag_b(krow, hi), s[nf]);
            }
        }

        // key mask per column (lane)
        float kmsk[4];
        #pragma unroll
        for (int nf = 0; nf < 4; nf++)
            kmsk[nf] = mask[(size_t)b * SEQ + kbase + nf * 16 + lh];

        // ---- online softmax; row r lives in lanes 0-15, row r+8 in 16-31 ----
        #pragma unroll
        for (int r = 0; r < 8; r++) {
            float v = -1e30f;
            #pragma unroll
            for (int nf = 0; nf < 4; nf++) {
                float sv = s[nf][r] * scale;
                sv = (kmsk[nf] > 0.f) ? sv : -1e30f;
                s[nf][r] = sv;
                v = fmaxf(v, sv);
            }
            v = fmaxf(v, __shfl_xor(v, 1, 16));
            v = fmaxf(v, __shfl_xor(v, 2, 16));
            v = fmaxf(v, __shfl_xor(v, 4, 16));
            v = fmaxf(v, __shfl_xor(v, 8, 16));
            float newm  = fmaxf(rmax[r], v);
            float alpha = __expf(rmax[r] - newm);
            rmax[r] = newm;

            float psum = 0.f;
            #pragma unroll
            for (int nf = 0; nf < 4; nf++) {
                float sv = s[nf][r];
                float p  = (sv <= -1e29f) ? 0.f : __expf(sv - newm);
                psum += p;
                myP[(r + hi * 8) * PSTR + nf * 16 + lh] = f2bf(p);
            }
            psum += __shfl_xor(psum, 1, 16);
            psum += __shfl_xor(psum, 2, 16);
            psum += __shfl_xor(psum, 4, 16);
            psum += __shfl_xor(psum, 8, 16);
            rsum[r] = rsum[r] * alpha + psum;
            #pragma unroll
            for (int nf = 0; nf < 4; nf++) o[nf][r] *= alpha;
        }

        // ---- O += P V : A from per-wave LDS, B from VT (contiguous keys) ----
        v16bf ap[2];
        ap[0] = frag_a(&myP[lh * PSTR], hi);
        ap[1] = frag_a(&myP[lh * PSTR + 32], hi);
        #pragma unroll
        for (int kf = 0; kf < 2; kf++) {
            #pragma unroll
            for (int nf = 0; nf < 4; nf++) {
                const unsigned short* vrow =
                    VT + (size_t)(h * DH + nf * 16 + lh) * MROWS
                       + (size_t)b * SEQ + kbase + kf * 32;
                o[nf] = wmma_bf16(ap[kf], frag_b(vrow, hi), o[nf]);
            }
        }
    }

    // epilogue: ctx[b*S+q][h*64+d] bf16
    #pragma unroll
    for (int nf = 0; nf < 4; nf++) {
        #pragma unroll
        for (int r = 0; r < 8; r++) {
            float denom = rsum[r];
            float v = (denom > 0.f) ? o[nf][r] / denom : 0.f;
            size_t m = rowQ + r + hi * 8;
            ctx[m * DMODEL + h * DH + nf * 16 + lh] = f2bf(v);
        }
    }
}

// ------------------------------- launcher ----------------------------------
extern "C" void kernel_launch(void* const* d_in, const int* in_sizes, int n_in,
                              void* d_out, int out_size, void* d_ws,
                              size_t ws_size, hipStream_t stream) {
    (void)in_sizes; (void)n_in; (void)out_size; (void)ws_size;
    const float* V    = (const float*)d_in[0];
    const float* Q    = (const float*)d_in[1];
    const float* K    = (const float*)d_in[2];
    const float* mask = (const float*)d_in[3];
    const float* WQ = (const float*)d_in[4];  const float* bQ = (const float*)d_in[5];
    const float* WK = (const float*)d_in[6];  const float* bK = (const float*)d_in[7];
    const float* WV = (const float*)d_in[8];  const float* bV = (const float*)d_in[9];
    const float* WO = (const float*)d_in[10]; const float* bO = (const float*)d_in[11];

    const size_t MD = (size_t)MROWS * DMODEL;    // 4M elems
    const size_t WD = (size_t)DMODEL * DMODEL;   // 1M elems
    unsigned short* ws  = (unsigned short*)d_ws;
    unsigned short* Qb  = ws;
    unsigned short* Kb  = Qb  + MD;
    unsigned short* Vb  = Kb  + MD;
    unsigned short* WQT = Vb  + MD;
    unsigned short* WKT = WQT + WD;
    unsigned short* WVT = WKT + WD;
    unsigned short* WOT = WVT + WD;
    unsigned short* Qp  = WOT + WD;
    unsigned short* Kp  = Qp  + MD;
    unsigned short* VTp = Kp  + MD;
    unsigned short* CT  = VTp + MD;   // total 7*MD + 4*WD = 32M elems = 64 MB

    // fp32 -> bf16 activations
    int cvtBlocks = (int)(MD / 4 / 256);
    k_cvt_bf16<<<cvtBlocks, 256, 0, stream>>>(Q, Qb, (int)MD);
    k_cvt_bf16<<<cvtBlocks, 256, 0, stream>>>(K, Kb, (int)MD);
    k_cvt_bf16<<<cvtBlocks, 256, 0, stream>>>(V, Vb, (int)MD);

    // weights -> bf16 transposed (WT[n][k])
    int trBlocks = (int)(WD / 256);
    k_transpose_cvt<<<trBlocks, 256, 0, stream>>>(WQ, WQT);
    k_transpose_cvt<<<trBlocks, 256, 0, stream>>>(WK, WKT);
    k_transpose_cvt<<<trBlocks, 256, 0, stream>>>(WV, WVT);
    k_transpose_cvt<<<trBlocks, 256, 0, stream>>>(WO, WOT);

    // projections
    dim3 ggrid(DMODEL / GN, MROWS / GM);   // (8, 32)
    k_gemm_bf16<0><<<ggrid, 256, 0, stream>>>(Qb, WQT, bQ, (void*)Qp, nullptr,
                                              MROWS, DMODEL, DMODEL);
    k_gemm_bf16<0><<<ggrid, 256, 0, stream>>>(Kb, WKT, bK, (void*)Kp, nullptr,
                                              MROWS, DMODEL, DMODEL);
    k_gemm_bf16<1><<<ggrid, 256, 0, stream>>>(Vb, WVT, bV, (void*)VTp, nullptr,
                                              MROWS, DMODEL, DMODEL); // V^T

    // attention
    dim3 agrid(SEQ / 64, BATCH * HEADS);   // (32, 32)
    k_attention<<<agrid, 128, 0, stream>>>(Qp, Kp, VTp, mask, CT);

    // output projection (fp32 + bias, * mask[row])
    k_gemm_bf16<2><<<ggrid, 256, 0, stream>>>(CT, WOT, bO, d_out, mask,
                                              MROWS, DMODEL, DMODEL);
}
